// XFormersCosineAttnProcessor_68015102099978
// MI455X (gfx1250) — matrix operand order repaired and checked by reading, not compiled, not run on hardware
//
#include <hip/hip_runtime.h>

// ---------------------------------------------------------------------------
// Cosine-sim attention block (XFormersCosineAttnProcessor) for MI455X/gfx1250
// B=4, T=2048, C=512, H=8, d=64.  All matrix math on v_wmma_f32_16x16x32_f16.
// All LDS fragment reads are contiguous 32B runs -> ds_load_b128 pairs.
// Prefetches use WGP scope (locality 3) since the working set is L2-resident.
// ---------------------------------------------------------------------------

typedef _Float16 v16h __attribute__((ext_vector_type(16)));
typedef _Float16 v8h  __attribute__((ext_vector_type(8)));
typedef _Float16 v4h  __attribute__((ext_vector_type(4)));
typedef float    v8f  __attribute__((ext_vector_type(8)));

static __device__ inline v8f v8f_zero() {
  v8f z = {0.f, 0.f, 0.f, 0.f, 0.f, 0.f, 0.f, 0.f};
  return z;
}

static __device__ inline v8f wmma_f16(v16h a, v16h b, v8f c) {
  // D = A(16x32 f16) * B(32x16 f16) + C(16x16 f32)
  return __builtin_amdgcn_wmma_f32_16x16x32_f16(
      /*neg_a=*/false, a, /*neg_b=*/false, b,
      /*c_mod=*/(short)0, c, /*reuse_a=*/false, /*reuse_b=*/false);
}

static __device__ inline v16h frag16(v8h lo, v8h hi) {
  union { v16h v; v8h h[2]; } u;
  u.h[0] = lo;
  u.h[1] = hi;
  return u.v;
}

// A-matrix fragment (16x32 f16) from row-major tile, leading dim ld (mult of 8).
// Contiguous k order: 16 contiguous halves per lane,
// with per-lane base row=(lane&15), +16 halves for lanes 16-31.
static __device__ inline v16h load_a_frag(const _Float16* __restrict__ tile, int ld) {
  const int lane = threadIdx.x & 31;
  const _Float16* p = tile + (lane & 15) * ld + ((lane >> 4) << 4);
  return frag16(*(const v8h*)p, *(const v8h*)(p + 8));
}

// B-matrix fragment (32x16 f16) from an N-major tile (tile[n][k], ld mult of 8):
// lane = N column, elements j = K = j + 16*(lane>=16) -> 16 contiguous halves.
static __device__ inline v16h load_bT_frag(const _Float16* __restrict__ tile, int ld) {
  const int lane = threadIdx.x & 31;
  const _Float16* p = tile + (lane & 15) * ld + ((lane >> 4) << 4);
  return frag16(*(const v8h*)p, *(const v8h*)(p + 8));
}

// Generic contiguous 16-half fragment at p (16B aligned).
static __device__ inline v16h load_contig16(const _Float16* __restrict__ p) {
  return frag16(*(const v8h*)p, *(const v8h*)(p + 8));
}

// ---------------------------------------------------------------------------
// 1) f32 -> f16 convert (vectorized x4)
// ---------------------------------------------------------------------------
__global__ __launch_bounds__(256) void k_f32_to_f16(const float* __restrict__ s,
                                                    _Float16* __restrict__ d, long n) {
  long i = ((long)blockIdx.x * 256 + threadIdx.x) * 4;
  if (i + 3 < n) {
    float4 x = *(const float4*)(s + i);
    v4h h = {(_Float16)x.x, (_Float16)x.y, (_Float16)x.z, (_Float16)x.w};
    *(v4h*)(d + i) = h;
  }
}

// ---------------------------------------------------------------------------
// 2) f16 GEMM, f16 out:  C[M,N] = A[M,K] * B[K,N]
//    block = 256 thr (8 waves), tile 128x64, K step 32, wave tile 32x32.
//    A staged row-major (ld 40); B staged N-major/transposed (ld 40).
// ---------------------------------------------------------------------------
__global__ __launch_bounds__(256) void k_gemm_f16(const _Float16* __restrict__ A,
                                                  const _Float16* __restrict__ Bm,
                                                  _Float16* __restrict__ Cm,
                                                  int M, int N, int K) {
  __shared__ _Float16 As[128 * 40];   // 128 rows x 32 k
  __shared__ _Float16 Bst[64 * 40];   // 64 cols  x 32 k  (transposed)
  const int tid = threadIdx.x;
  const int wave = tid >> 5;
  const int wm = wave >> 1;           // 0..3
  const int wn = wave & 1;            // 0..1
  const long rowBase = (long)blockIdx.x * 128;
  const int colBase = blockIdx.y * 64;

  v8f acc[2][2];
#pragma unroll
  for (int i = 0; i < 2; ++i)
#pragma unroll
    for (int j = 0; j < 2; ++j) acc[i][j] = v8f_zero();

  const int ar = tid >> 1, ac = (tid & 1) * 16;   // A staging coords
  const int br = tid >> 3, bc = (tid & 7) * 8;    // B staging coords

  for (int k0 = 0; k0 < K; k0 += 32) {
    {  // stage A tile: 128x32 halves, 32B per thread (vector copy)
      const _Float16* src = A + (rowBase + ar) * (long)K + k0 + ac;
      _Float16* dst = As + ar * 40 + ac;
      *(v8h*)(dst) = *(const v8h*)(src);
      *(v8h*)(dst + 8) = *(const v8h*)(src + 8);
    }
    {  // stage B tile transposed: read 8 contiguous halves of row k, scatter
      const v8h v = *(const v8h*)(Bm + (long)(k0 + br) * N + colBase + bc);
#pragma unroll
      for (int u = 0; u < 8; ++u) Bst[(bc + u) * 40 + br] = v[u];
    }
    if (k0 + 32 < K) {  // WGP-scope prefetch of next K-step into near caches
      __builtin_prefetch(A + (rowBase + ar) * (long)K + k0 + 32 + ac, 0, 3);
      __builtin_prefetch(Bm + (long)(k0 + 32 + br) * N + colBase + bc, 0, 3);
    }
    __syncthreads();

    const v16h a0 = load_a_frag(As + (wm * 32 + 0) * 40, 40);
    const v16h a1 = load_a_frag(As + (wm * 32 + 16) * 40, 40);
    const v16h b0 = load_bT_frag(Bst + (wn * 32 + 0) * 40, 40);
    const v16h b1 = load_bT_frag(Bst + (wn * 32 + 16) * 40, 40);
    acc[0][0] = wmma_f16(a0, b0, acc[0][0]);
    acc[0][1] = wmma_f16(a0, b1, acc[0][1]);
    acc[1][0] = wmma_f16(a1, b0, acc[1][0]);
    acc[1][1] = wmma_f16(a1, b1, acc[1][1]);
    __syncthreads();
  }

  const int lane = tid & 31;
  const int laneHi = lane >> 4, lanelo = lane & 15;
#pragma unroll
  for (int mi = 0; mi < 2; ++mi)
#pragma unroll
    for (int ni = 0; ni < 2; ++ni)
#pragma unroll
      for (int r = 0; r < 8; ++r) {
        const long row = rowBase + wm * 32 + mi * 16 + r + laneHi * 8;
        const int col = colBase + wn * 32 + ni * 16 + lanelo;
        Cm[row * N + col] = (_Float16)acc[mi][ni][r];
      }
}

// ---------------------------------------------------------------------------
// 3) pixel-norm over token axis: inv[mb][c] = rsqrt(mean_t x^2 + eps)
// ---------------------------------------------------------------------------
__global__ __launch_bounds__(256) void k_colnorm_inv(const _Float16* __restrict__ qkv,
                                                     float* __restrict__ inv,
                                                     int T, int C) {
  const int mb = blockIdx.x;
  const int cbase = blockIdx.y * 64;
  const int ch = threadIdx.x & 63;
  const int tg = threadIdx.x >> 6;  // 0..3
  const _Float16* base = qkv + (long)mb * T * C + cbase + ch;
  float s = 0.f;
  for (int t = tg; t < T; t += 4) {
    const float x = (float)base[(long)t * C];
    s += x * x;
  }
  __shared__ float red[4][64];
  red[tg][ch] = s;
  __syncthreads();
  if (tg == 0) {
    const float tot = red[0][ch] + red[1][ch] + red[2][ch] + red[3][ch];
    inv[(long)mb * C + cbase + ch] = rsqrtf(tot / (float)T + 1e-4f);
  }
}

__global__ __launch_bounds__(256) void k_apply_norm(_Float16* __restrict__ qkv,
                                                    const float* __restrict__ inv,
                                                    int T, int C) {
  const long row = blockIdx.x;               // [0, 3*B*T)
  const float* ip = inv + (row / T) * C;
  _Float16* p = qkv + row * C;
  const int c = threadIdx.x * 2;
  const float x0 = (float)p[c] * ip[c];
  const float x1 = (float)p[c + 1] * ip[c + 1];
  p[c] = (_Float16)x0;
  p[c + 1] = (_Float16)x1;
}

// ---------------------------------------------------------------------------
// 4) flash attention: grid (B*H, T/128); 8 waves x 16 query rows each.
//    key tile = 32, online softmax per wave (no cross-wave reductions).
//    K staged row-major (QK^T frags contiguous); V staged d-major so that
//    PV B-fragments are contiguous.
// ---------------------------------------------------------------------------
__global__ __launch_bounds__(256) void k_attention(const _Float16* __restrict__ qh,
                                                   const _Float16* __restrict__ kh,
                                                   const _Float16* __restrict__ vh,
                                                   _Float16* __restrict__ oh,
                                                   int H, int T, int C) {
  const int b = blockIdx.x / H;
  const int h = blockIdx.x % H;
  const int wave = threadIdx.x >> 5;
  const int lane = threadIdx.x & 31;
  const int laneHi = lane >> 4, lanelo = lane & 15;
  const long qrow0 = (long)blockIdx.y * 128 + wave * 16;

  const _Float16* qbase = qh + (long)b * T * C + h * 64;
  const _Float16* kbase = kh + (long)b * T * C + h * 64;
  const _Float16* vbase = vh + (long)b * T * C + h * 64;

  __shared__ _Float16 Kt[32 * 72];        // 32 keys x 64 d   (row-major)
  __shared__ _Float16 Vst[64 * 40];       // 64 d    x 32 keys (transposed)
  __shared__ _Float16 Pt[8][16 * 40];     // per-wave P transpose staging

  // Q fragments: 16 rows x 64 d -> two A frags (d chunks of 32), contiguous
  v16h aQ[2];
#pragma unroll
  for (int ch = 0; ch < 2; ++ch) {
    const _Float16* p = qbase + (qrow0 + lanelo) * C + ch * 32 + (laneHi << 4);
    aQ[ch] = frag16(*(const v8h*)p, *(const v8h*)(p + 8));
  }

  v8f o[4];
#pragma unroll
  for (int dt = 0; dt < 4; ++dt) o[dt] = v8f_zero();
  float mrun[8], lrun[8];
#pragma unroll
  for (int r = 0; r < 8; ++r) { mrun[r] = -1e30f; lrun[r] = 0.f; }

  const float scale = 0.125f;  // d^-0.5, d=64
  const int sr = threadIdx.x >> 3, sc = (threadIdx.x & 7) * 8;  // staging coords

  for (int kt = 0; kt < T; kt += 32) {
    {  // stage K row-major (vector), V transposed (scatter)
      *(v8h*)(Kt + sr * 72 + sc) = *(const v8h*)(kbase + (long)(kt + sr) * C + sc);
      const v8h v = *(const v8h*)(vbase + (long)(kt + sr) * C + sc);
#pragma unroll
      for (int u = 0; u < 8; ++u) Vst[(sc + u) * 40 + sr] = v[u];
    }
    if (kt + 32 < T) {  // WGP-scope prefetch of next key tile
      __builtin_prefetch(kbase + (long)(kt + 32 + sr) * C + sc, 0, 3);
      __builtin_prefetch(vbase + (long)(kt + 32 + sr) * C + sc, 0, 3);
    }
    __syncthreads();

    // S = Q * K^T : two 16x16 tiles (keys 0-15 and 16-31)
    v8f s[2];
    s[0] = v8f_zero();
    s[1] = v8f_zero();
#pragma unroll
    for (int nt = 0; nt < 2; ++nt) {
#pragma unroll
      for (int ch = 0; ch < 2; ++ch) {
        // B frag of K^T: lane = key column, 16 contiguous d elements
        const v16h bf =
            load_contig16(Kt + (nt * 16 + lanelo) * 72 + ch * 32 + (laneHi << 4));
        s[nt] = wmma_f16(aQ[ch], bf, s[nt]);
      }
    }

    // online softmax (rows live in 16-lane halves; shfl_xor masks 1..8)
#pragma unroll
    for (int r = 0; r < 8; ++r) {
      const float x0 = s[0][r] * scale;
      const float x1 = s[1][r] * scale;
      float mx = fmaxf(x0, x1);
#pragma unroll
      for (int off = 8; off >= 1; off >>= 1) mx = fmaxf(mx, __shfl_xor(mx, off, 32));
      const float mnew = fmaxf(mrun[r], mx);
      const float p0 = __expf(x0 - mnew);
      const float p1 = __expf(x1 - mnew);
      float ls = p0 + p1;
#pragma unroll
      for (int off = 8; off >= 1; off >>= 1) ls += __shfl_xor(ls, off, 32);
      const float alpha = __expf(mrun[r] - mnew);
      lrun[r] = lrun[r] * alpha + ls;
      mrun[r] = mnew;
#pragma unroll
      for (int dt = 0; dt < 4; ++dt) o[dt][r] *= alpha;
      s[0][r] = p0;
      s[1][r] = p1;
    }

    // transpose P (C/D layout -> A layout) through per-wave LDS tile
    _Float16* pt = &Pt[wave][0];
#pragma unroll
    for (int r = 0; r < 8; ++r) {
      const int row = r + laneHi * 8;
      pt[row * 40 + lanelo] = (_Float16)s[0][r];
      pt[row * 40 + 16 + lanelo] = (_Float16)s[1][r];
    }
    // same-wave LDS ops are in-order: safe to regather immediately
    const v16h aP = load_a_frag(pt, 40);

    // O += P @ V : four 16-wide d tiles, contiguous frags from d-major Vst
#pragma unroll
    for (int dt = 0; dt < 4; ++dt) {
      const v16h bf = load_bT_frag(Vst + (dt * 16) * 40, 40);
      o[dt] = wmma_f16(aP, bf, o[dt]);
    }
    __syncthreads();
  }

  // epilogue: O / l, store f16 [B,T,C]
  _Float16* obase = oh + (long)b * T * C + h * 64;
#pragma unroll
  for (int r = 0; r < 8; ++r) {
    const float li = 1.0f / lrun[r];
    const long trow = qrow0 + r + laneHi * 8;
#pragma unroll
    for (int dt = 0; dt < 4; ++dt)
      obase[trow * C + dt * 16 + lanelo] = (_Float16)(o[dt][r] * li);
  }
}

// ---------------------------------------------------------------------------
// 5) output GEMM with fused bias + residual, f32 out
// ---------------------------------------------------------------------------
__global__ __launch_bounds__(256) void k_gemm_out(const _Float16* __restrict__ A,
                                                  const _Float16* __restrict__ Bm,
                                                  const float* __restrict__ bias,
                                                  const float* __restrict__ resid,
                                                  float* __restrict__ out,
                                                  int M, int N, int K) {
  __shared__ _Float16 As[128 * 40];
  __shared__ _Float16 Bst[64 * 40];
  const int tid = threadIdx.x;
  const int wave = tid >> 5;
  const int wm = wave >> 1;
  const int wn = wave & 1;
  const long rowBase = (long)blockIdx.x * 128;
  const int colBase = blockIdx.y * 64;

  v8f acc[2][2];
#pragma unroll
  for (int i = 0; i < 2; ++i)
#pragma unroll
    for (int j = 0; j < 2; ++j) acc[i][j] = v8f_zero();

  const int ar = tid >> 1, ac = (tid & 1) * 16;
  const int br = tid >> 3, bc = (tid & 7) * 8;

  for (int k0 = 0; k0 < K; k0 += 32) {
    {
      const _Float16* src = A + (rowBase + ar) * (long)K + k0 + ac;
      _Float16* dst = As + ar * 40 + ac;
      *(v8h*)(dst) = *(const v8h*)(src);
      *(v8h*)(dst + 8) = *(const v8h*)(src + 8);
    }
    {
      const v8h v = *(const v8h*)(Bm + (long)(k0 + br) * N + colBase + bc);
#pragma unroll
      for (int u = 0; u < 8; ++u) Bst[(bc + u) * 40 + br] = v[u];
    }
    if (k0 + 32 < K) {
      __builtin_prefetch(A + (rowBase + ar) * (long)K + k0 + 32 + ac, 0, 3);
      __builtin_prefetch(Bm + (long)(k0 + 32 + br) * N + colBase + bc, 0, 3);
    }
    __syncthreads();
    const v16h a0 = load_a_frag(As + (wm * 32 + 0) * 40, 40);
    const v16h a1 = load_a_frag(As + (wm * 32 + 16) * 40, 40);
    const v16h b0 = load_bT_frag(Bst + (wn * 32 + 0) * 40, 40);
    const v16h b1 = load_bT_frag(Bst + (wn * 32 + 16) * 40, 40);
    acc[0][0] = wmma_f16(a0, b0, acc[0][0]);
    acc[0][1] = wmma_f16(a0, b1, acc[0][1]);
    acc[1][0] = wmma_f16(a1, b0, acc[1][0]);
    acc[1][1] = wmma_f16(a1, b1, acc[1][1]);
    __syncthreads();
  }

  const int lane = tid & 31;
  const int laneHi = lane >> 4, lanelo = lane & 15;
#pragma unroll
  for (int mi = 0; mi < 2; ++mi)
#pragma unroll
    for (int ni = 0; ni < 2; ++ni)
#pragma unroll
      for (int r = 0; r < 8; ++r) {
        const long row = rowBase + wm * 32 + mi * 16 + r + laneHi * 8;
        const int col = colBase + wn * 32 + ni * 16 + lanelo;
        out[row * N + col] = acc[mi][ni][r] + bias[col] + resid[row * N + col];
      }
}

// ---------------------------------------------------------------------------
// launch
// ---------------------------------------------------------------------------
extern "C" void kernel_launch(void* const* d_in, const int* in_sizes, int n_in,
                              void* d_out, int out_size, void* d_ws, size_t ws_size,
                              hipStream_t stream) {
  (void)in_sizes; (void)n_in; (void)out_size;
  const float* hs = (const float*)d_in[0];
  const float* Wq = (const float*)d_in[1];
  const float* Wk = (const float*)d_in[2];
  const float* Wv = (const float*)d_in[3];
  const float* Wo = (const float*)d_in[4];
  const float* bo = (const float*)d_in[5];

  const int B = 4, T = 2048, C = 512, H = 8;
  const int M = B * T;                      // 8192
  const long BTC = (long)M * C;             // 4,194,304
  const long CC = (long)C * C;              // 262,144

  // workspace layout (f16 unless noted); total ~42 MB
  _Float16* hs_h   = (_Float16*)d_ws;       // BTC
  _Float16* wq_h   = hs_h + BTC;            // CC
  _Float16* wk_h   = wq_h + CC;
  _Float16* wv_h   = wk_h + CC;
  _Float16* wo_h   = wv_h + CC;
  _Float16* qkv_h  = wo_h + CC;             // 3*BTC  ([3][B,T,C])
  _Float16* attn_h = qkv_h + 3 * BTC;       // BTC
  float*    inv    = (float*)(attn_h + BTC);// 3*B*C floats
  (void)ws_size;

  // 1) convert inputs to f16
  k_f32_to_f16<<<(int)(BTC / (256 * 4)), 256, 0, stream>>>(hs, hs_h, BTC);
  k_f32_to_f16<<<(int)(CC / (256 * 4)), 256, 0, stream>>>(Wq, wq_h, CC);
  k_f32_to_f16<<<(int)(CC / (256 * 4)), 256, 0, stream>>>(Wk, wk_h, CC);
  k_f32_to_f16<<<(int)(CC / (256 * 4)), 256, 0, stream>>>(Wv, wv_h, CC);
  k_f32_to_f16<<<(int)(CC / (256 * 4)), 256, 0, stream>>>(Wo, wo_h, CC);

  // 2) Q/K/V projections
  dim3 ggrid(M / 128, C / 64);
  k_gemm_f16<<<ggrid, 256, 0, stream>>>(hs_h, wq_h, qkv_h + 0 * BTC, M, C, C);
  k_gemm_f16<<<ggrid, 256, 0, stream>>>(hs_h, wk_h, qkv_h + 1 * BTC, M, C, C);
  k_gemm_f16<<<ggrid, 256, 0, stream>>>(hs_h, wv_h, qkv_h + 2 * BTC, M, C, C);

  // 3) pixel-norm over token axis
  k_colnorm_inv<<<dim3(3 * B, C / 64), 256, 0, stream>>>(qkv_h, inv, T, C);
  k_apply_norm<<<3 * B * T, 256, 0, stream>>>(qkv_h, inv, T, C);

  // 4) attention
  k_attention<<<dim3(B * H, T / 128), 256, 0, stream>>>(
      qkv_h, qkv_h + BTC, qkv_h + 2 * BTC, attn_h, H, T, C);

  // 5) output projection + bias + residual
  k_gemm_out<<<ggrid, 256, 0, stream>>>(attn_h, wo_h, bo, hs, (float*)d_out, M, C, C);
}
